// D2A_12086037971054
// MI455X (gfx1250) — matrix operand order repaired and checked by reading, not compiled, not run on hardware
//
#include <hip/hip_runtime.h>
#include <math.h>

// ---------------------------------------------------------------------------
// out = x + x @ Att  with Att = colnorm(mask(MLP(DistM)))  [M=65536,K=N=1024]
// GEMM on v_wmma_f32_16x16x32_bf16, f32 accumulate. Block tile 128(M)x64(N),
// BK=64: 8 waves stacked in M; per staged buffer each wave runs 2 K-substeps
// x 4 N-subtiles = 8 WMMAs. sched_group_barrier pins the schedule so one
// B-fragment ds_load_b128 pair stays in flight ahead of every WMMA (partial
// dscnt waits, back-to-back matrix issue). Double-buffered LDS staging
// overlaps global->LDS of tile kb+64 with the math on tile kb.
// AttT (bf16, N-major) lives in d_ws (2 MB).
// ---------------------------------------------------------------------------

typedef __attribute__((ext_vector_type(16))) __bf16 v16bf;
typedef __attribute__((ext_vector_type(8)))  float  v8f;
typedef __attribute__((ext_vector_type(4)))  __bf16 bf16x4;

#define F_DIM 1024
#define H_DIM 64
#define EPSV  1e-7f
#define BM    128
#define BN    64
#define BK    64
#define LDA   72   // padded LDS row stride (bf16 elems) -> conflict-free b128
#define LDB   72

#define SGB_DSREAD 0x100
#define SGB_MMA    0x008

__device__ __forceinline__ float mlp_scalar(float z,
                                            const float* __restrict__ W1,
                                            const float* __restrict__ b1,
                                            const float* __restrict__ W2,
                                            float b2v) {
  float acc = b2v;
#pragma unroll 8
  for (int j = 0; j < H_DIM; ++j)
    acc = fmaf(W2[j], tanhf(fmaf(z, W1[j], b1[j])), acc);
  return acc;
}

// One block per Att column n: eval MLP, mask diag, reduce column sum,
// write normalized column as row n of AttT (bf16, K-contiguous).
__global__ void att_kernel(const float* __restrict__ DistM,
                           const float* __restrict__ W1,
                           const float* __restrict__ b1,
                           const float* __restrict__ W2,
                           const float* __restrict__ b2,
                           unsigned short* __restrict__ AttT_u) {
  __shared__ float sW1[H_DIM], sB1[H_DIM], sW2[H_DIM];
  __shared__ float red[256];
  __bf16* AttT = reinterpret_cast<__bf16*>(AttT_u);
  const int n = blockIdx.x;
  const int t = threadIdx.x;
  if (t < H_DIM) { sW1[t] = W1[t]; sB1[t] = b1[t]; sW2[t] = W2[t]; }
  __syncthreads();
  const float b2v = b2[0];
  float vals[4];
  float lsum = 0.f;
#pragma unroll
  for (int i = 0; i < 4; ++i) {
    const int k = t * 4 + i;
    float v = mlp_scalar(DistM[(size_t)k * F_DIM + n], sW1, sB1, sW2, b2v);
    if (k == n) v = 0.f;                 // off-diagonal mask
    vals[i] = v;
    lsum += v;
  }
  red[t] = lsum;
  __syncthreads();
  for (int s = 128; s > 0; s >>= 1) {
    if (t < s) red[t] += red[t + s];
    __syncthreads();
  }
  const float inv = 1.0f / (red[0] + EPSV);
  bf16x4 o;
#pragma unroll
  for (int i = 0; i < 4; ++i) o[i] = (__bf16)(vals[i] * inv);
  *reinterpret_cast<bf16x4*>(&AttT[(size_t)n * F_DIM + t * 4]) = o;
}

// Single-wave keypoint monotonicity penalty.
__global__ void penalty_kernel(const float* __restrict__ W1,
                               const float* __restrict__ b1,
                               const float* __restrict__ W2,
                               const float* __restrict__ b2,
                               float* __restrict__ out) {
  __shared__ float kv[32];
  const int t = threadIdx.x;
  kv[t] = mlp_scalar((float)t * (1.0f / 31.0f), W1, b1, W2, b2[0]);
  __syncthreads();
  if (t == 0) {
    float p = 0.f;
    for (int i = 1; i < 32; ++i) {
      float d = kv[i] - kv[i - 1];
      p += fabsf(d) + d;
    }
    out[0] = 0.5f * 0.1f * (p * (1.0f / 31.0f));
  }
}

__global__ void __launch_bounds__(256, 2)
gemm_kernel(const float* __restrict__ x,
            const unsigned short* __restrict__ AttT_u,
            float* __restrict__ out) {
  __shared__ __align__(16) __bf16 lA[2][BM * LDA];   // 2 x 18 KB
  __shared__ __align__(16) __bf16 lB[2][BN * LDB];   // 2 x  9 KB
  const __bf16* AttT = reinterpret_cast<const __bf16*>(AttT_u);

  const int t    = threadIdx.x;
  const int wave = t >> 5;
  const int lane = t & 31;
  const int nblk = blockIdx.x * BN;
  const int mblk = blockIdx.y * BM;

  // staging assignments (256 threads)
  const int mrow  = t >> 1;          // 0..127 : A row, 32 floats / thread
  const int khalf = (t & 1) * 32;    // 0|32   : A k-half
  const int bn    = t >> 2;          // 0..63  : AttT row (output column)
  const int bpart = (t & 3) * 16;    // 0..48  : 16 bf16 / thread

  // fragment lane mapping (ISA 7.12.2)
  const int fm = lane & 15;          // A: M row / B: N column
  const int fg = lane >> 4;          // half-select

  const float*  aSrc = x    + (size_t)(mblk + mrow) * F_DIM + khalf;
  const __bf16* bSrc = AttT + (size_t)(nblk + bn)  * F_DIM + bpart;

  auto stage = [&](int buf, int kb) {
    const float4* a4 = reinterpret_cast<const float4*>(aSrc + kb);
    __bf16* dA = &lA[buf][mrow * LDA + khalf];
#pragma unroll
    for (int q = 0; q < 8; ++q) {
      float4 f = a4[q];
      dA[q * 4 + 0] = (__bf16)f.x;
      dA[q * 4 + 1] = (__bf16)f.y;
      dA[q * 4 + 2] = (__bf16)f.z;
      dA[q * 4 + 3] = (__bf16)f.w;
    }
    const uint4* b4 = reinterpret_cast<const uint4*>(bSrc + kb);
    uint4* dB = reinterpret_cast<uint4*>(&lB[buf][bn * LDB + bpart]);
    dB[0] = b4[0];
    dB[1] = b4[1];
  };

  v8f acc[4];
#pragma unroll
  for (int j = 0; j < 4; ++j) acc[j] = (v8f){0.f,0.f,0.f,0.f,0.f,0.f,0.f,0.f};

  stage(0, 0);
  __syncthreads();

  union Frag { v16bf v; uint4 q[2]; };

  int buf = 0;
  for (int kb = 0; kb < F_DIM; kb += BK, buf ^= 1) {
    if (kb + BK < F_DIM) {
      stage(buf ^ 1, kb + BK);                        // fill other buffer
      if (kb + 2 * BK < F_DIM)
        __builtin_prefetch(aSrc + kb + 2 * BK, 0, 1); // global_prefetch_b8
    }
#pragma unroll
    for (int ks = 0; ks < 2; ++ks) {                  // two K=32 substeps
      // batch-load A fragment + all 4 B fragments, then chain 4 WMMAs
      Frag fa;
      const __bf16* ap = &lA[buf][(wave * 16 + fm) * LDA + ks * 32];
      fa.q[0] = *reinterpret_cast<const uint4*>(ap + fg * 8);
      fa.q[1] = *reinterpret_cast<const uint4*>(ap + 16 + fg * 8);
      Frag fb[4];
#pragma unroll
      for (int j = 0; j < 4; ++j) {
        const __bf16* bp = &lB[buf][(j * 16 + fm) * LDB + ks * 32 + fg * 8];
        fb[j].q[0] = *reinterpret_cast<const uint4*>(bp);
        fb[j].q[1] = *reinterpret_cast<const uint4*>(bp + 16);
      }
#pragma unroll
      for (int j = 0; j < 4; ++j)
        acc[j] = __builtin_amdgcn_wmma_f32_16x16x32_bf16(
            false, fa.v, false, fb[j].v, (short)0, acc[j], false, false);

      // Pin schedule: fa+fb0+fb1 (6 DS reads) up front, then keep one B
      // fragment (2 DS reads) in flight ahead of each WMMA.
      __builtin_amdgcn_sched_group_barrier(SGB_DSREAD, 6, 0);
      __builtin_amdgcn_sched_group_barrier(SGB_MMA,    1, 0);
      __builtin_amdgcn_sched_group_barrier(SGB_DSREAD, 2, 0);
      __builtin_amdgcn_sched_group_barrier(SGB_MMA,    1, 0);
      __builtin_amdgcn_sched_group_barrier(SGB_DSREAD, 2, 0);
      __builtin_amdgcn_sched_group_barrier(SGB_MMA,    1, 0);
      __builtin_amdgcn_sched_group_barrier(SGB_MMA,    1, 0);
    }
    __syncthreads();
  }

  // epilogue: out = x + acc  (C layout: n = l&15, m = 8*(l>>4)+i)
  const int mbase = mblk + wave * 16 + fg * 8;
#pragma unroll
  for (int j = 0; j < 4; ++j) {
    const int n = nblk + j * 16 + fm;
#pragma unroll
    for (int i = 0; i < 8; ++i) {
      const size_t idx = (size_t)(mbase + i) * F_DIM + n;
      out[idx] = x[idx] + acc[j][i];
    }
  }
}

extern "C" void kernel_launch(void* const* d_in, const int* in_sizes, int n_in,
                              void* d_out, int out_size, void* d_ws, size_t ws_size,
                              hipStream_t stream) {
  (void)n_in; (void)out_size; (void)ws_size;
  const float* x     = (const float*)d_in[0];
  const float* DistM = (const float*)d_in[1];
  const float* W1    = (const float*)d_in[2];
  const float* b1    = (const float*)d_in[3];
  const float* W2    = (const float*)d_in[4];
  const float* b2    = (const float*)d_in[5];
  float*          out  = (float*)d_out;
  unsigned short* AttT = (unsigned short*)d_ws;   // 1024*1024 bf16 = 2 MB

  const int Mrows = in_sizes[0] / F_DIM;          // 65536

  att_kernel<<<F_DIM, 256, 0, stream>>>(DistM, W1, b1, W2, b2, AttT);
  penalty_kernel<<<1, 32, 0, stream>>>(W1, b1, W2, b2,
                                       out + (size_t)Mrows * F_DIM);
  dim3 grid(F_DIM / BN, Mrows / BM);              // 16 x 512 blocks
  gemm_kernel<<<grid, 256, 0, stream>>>(x, AttT, out);
}
